// Mamba2Block_55456617726038
// MI455X (gfx1250) — matrix operand order repaired and checked by reading, not compiled, not run on hardware
//
#include <hip/hip_runtime.h>

// ---------------------------------------------------------------------------
// Mamba2 block forward for gfx1250 (MI455X).
// GEMMs: v_wmma_f32_16x16x32_bf16, 128x128 block tile, 8 waves (32x64 each),
// double-buffered LDS filled via async global->LDS loads when available.
// ---------------------------------------------------------------------------

typedef __attribute__((ext_vector_type(16))) __bf16 v16bf;
typedef __attribute__((ext_vector_type(8)))  float  v8f;
typedef int i32x4v __attribute__((vector_size(16)));

#define D_MODEL   1024
#define D_INNER   2048
#define D_STATE   128
#define HEAD_DIM  64
#define N_HEADS   32
#define D_PROJ    4384   // 2*2048 + 2*128 + 32
#define SEQ_L     2048
#define BATCH     2
#define M_TOK     (BATCH * SEQ_L)   // 4096 rows

#define AS1 __attribute__((address_space(1)))
#define AS3 __attribute__((address_space(3)))

#if defined(__has_builtin)
#if __has_builtin(__builtin_amdgcn_global_load_async_to_lds_b128) && \
    __has_builtin(__builtin_amdgcn_s_wait_asynccnt)
#define USE_ASYNC_LDS 1
#endif
#endif
#ifndef USE_ASYNC_LDS
#define USE_ASYNC_LDS 0
#endif

__device__ __forceinline__ void cp16_g2l(const unsigned short* g,
                                         unsigned short* l) {
#if USE_ASYNC_LDS
    __builtin_amdgcn_global_load_async_to_lds_b128(
        (AS1 i32x4v*)g, (AS3 i32x4v*)l, 0, 0);
#else
    *(uint4*)l = *(const uint4*)g;
#endif
}

__device__ __forceinline__ void wait_async_fills() {
#if USE_ASYNC_LDS
    __builtin_amdgcn_s_wait_asynccnt(0);
#endif
}

__device__ __forceinline__ unsigned short f2bfu(float f) {
    unsigned u = __builtin_bit_cast(unsigned, f);
    unsigned r = u + 0x7FFFu + ((u >> 16) & 1u);   // round-to-nearest-even
    return (unsigned short)(r >> 16);
}

__device__ __forceinline__ float siluf(float v) {
    return v / (1.0f + expf(-v));
}

// --------------------------- f32 -> bf16 convert ---------------------------
__global__ void f32_to_bf16_kernel(const float* __restrict__ src,
                                   unsigned short* __restrict__ dst, int n) {
    int i = blockIdx.x * 256 + threadIdx.x;
    if (i < n) dst[i] = f2bfu(src[i]);
}

// ------------------------------- RMSNorm -----------------------------------
__global__ __launch_bounds__(256) void rmsnorm_kernel(
    const float* __restrict__ x, const float* __restrict__ w,
    unsigned short* __restrict__ xn_bf) {
    int m = blockIdx.x;
    int tid = threadIdx.x;
    const float* row = x + (size_t)m * D_MODEL;
    float v[4];
    float s = 0.f;
#pragma unroll
    for (int i = 0; i < 4; ++i) {
        v[i] = row[tid + 256 * i];
        s += v[i] * v[i];
    }
    __shared__ float red[256];
    red[tid] = s;
    __syncthreads();
    for (int off = 128; off > 0; off >>= 1) {
        if (tid < off) red[tid] += red[tid + off];
        __syncthreads();
    }
    float scale = rsqrtf(red[0] * (1.0f / D_MODEL) + 1e-6f);
#pragma unroll
    for (int i = 0; i < 4; ++i) {
        int j = tid + 256 * i;
        xn_bf[(size_t)m * D_MODEL + j] = f2bfu(v[i] * scale * w[j]);
    }
}

// ------------------------------ WMMA GEMM ----------------------------------
// C[M x ldc](f32) = A[M x K](bf16) * W[Nact x K](bf16)^T  (+ residual)
// block tile 128(M) x 128(N); 8 waves: waveM in 0..3 (32 rows), waveN in 0..1
// (64 cols); each wave: 2x4 wmma tiles (8 accumulators).
union AFrag { uint4 q[2]; v16bf v; };

__global__ __launch_bounds__(256) void wmma_gemm_kernel(
    const unsigned short* __restrict__ A,
    const unsigned short* __restrict__ W,
    float* __restrict__ C,
    const float* __restrict__ residual,
    int M, int Nact, int K, int ldc) {
    __shared__ __align__(32) unsigned short sA[2][128 * 32];
    __shared__ __align__(32) unsigned short sB[2][128 * 32];

    int tid  = threadIdx.x;
    int lane = tid & 31;
    int wv   = tid >> 5;
    int waveM = wv & 3;          // M offset 32*waveM
    int waveN = wv >> 2;         // N offset 64*waveN
    int bM = blockIdx.y * 128;
    int bN = blockIdx.x * 128;

    v8f acc[8];
#pragma unroll
    for (int i = 0; i < 8; ++i) acc[i] = (v8f){};

    int g = lane >> 4;           // half-lane group
    int r = lane & 15;

    // staging geometry: tile is 128 rows x 32 bf16 = 512 chunks of 16B;
    // 256 threads -> 2 chunks each; chunk c: row = c>>2, col-chunk = c&3
    int ksteps = K >> 5;

    // ---- fill buffer 0 ----
    {
#pragma unroll
        for (int i = 0; i < 2; ++i) {
            int c = tid + i * 256, row = c >> 2, cc = c & 3;
            cp16_g2l(&A[(size_t)(bM + row) * K + cc * 8],
                     &sA[0][row * 32 + cc * 8]);
        }
#pragma unroll
        for (int i = 0; i < 2; ++i) {
            int c = tid + i * 256, row = c >> 2, cc = c & 3;
            int n = bN + row;
            if (n >= Nact) n = Nact - 1;   // clamp; OOB cols never stored
            cp16_g2l(&W[(size_t)n * K + cc * 8], &sB[0][row * 32 + cc * 8]);
        }
    }
    wait_async_fills();
    __syncthreads();

    for (int ks = 0; ks < ksteps; ++ks) {
        int cur = ks & 1;
        // ---- fill next buffer while computing from current ----
        if (ks + 1 < ksteps) {
            int nxt = cur ^ 1;
            int k1  = (ks + 1) << 5;
#pragma unroll
            for (int i = 0; i < 2; ++i) {
                int c = tid + i * 256, row = c >> 2, cc = c & 3;
                cp16_g2l(&A[(size_t)(bM + row) * K + k1 + cc * 8],
                         &sA[nxt][row * 32 + cc * 8]);
            }
#pragma unroll
            for (int i = 0; i < 2; ++i) {
                int c = tid + i * 256, row = c >> 2, cc = c & 3;
                int n = bN + row;
                if (n >= Nact) n = Nact - 1;
                cp16_g2l(&W[(size_t)n * K + k1 + cc * 8],
                         &sB[nxt][row * 32 + cc * 8]);
            }
        }
        if (ks + 2 < ksteps) {
            // cache hint for the tile after next (global_prefetch_b8)
            __builtin_prefetch(&A[(size_t)(bM + (tid >> 2)) * K + (ks << 5) + 64],
                               0, 1);
        }

        // ---- fragments per documented WMMA VGPR layouts ----
        v16bf afrag[2], bfrag[4];
#pragma unroll
        for (int tm = 0; tm < 2; ++tm) {
            int base = (waveM * 32 + tm * 16 + r) * 32 + g * 8;
            AFrag f;
            f.q[0] = *(const uint4*)&sA[cur][base];       // K = g*8 + 0..7
            f.q[1] = *(const uint4*)&sA[cur][base + 16];  // K = 16+g*8 + 0..7
            afrag[tm] = f.v;
        }
#pragma unroll
        for (int tn = 0; tn < 4; ++tn) {
            int base = (waveN * 64 + tn * 16 + r) * 32 + g * 8;
            AFrag f;
            f.q[0] = *(const uint4*)&sB[cur][base];       // K = g*8 + 0..7
            f.q[1] = *(const uint4*)&sB[cur][base + 16];  // K = 16+g*8 + 0..7
            bfrag[tn] = f.v;
        }
#pragma unroll
        for (int tm = 0; tm < 2; ++tm)
#pragma unroll
            for (int tn = 0; tn < 4; ++tn)
                acc[tm * 4 + tn] = __builtin_amdgcn_wmma_f32_16x16x32_bf16(
                    false, afrag[tm], false, bfrag[tn],
                    (short)0, acc[tm * 4 + tn], false, false);

        wait_async_fills();
        __syncthreads();
    }

    // ---- store: lane holds col r, rows g*8 + j ----
#pragma unroll
    for (int tm = 0; tm < 2; ++tm) {
#pragma unroll
        for (int tn = 0; tn < 4; ++tn) {
            int col = bN + waveN * 64 + tn * 16 + r;
            if (col < Nact) {
                v8f a = acc[tm * 4 + tn];
#pragma unroll
                for (int j = 0; j < 8; ++j) {
                    int row = bM + waveM * 32 + tm * 16 + g * 8 + j;
                    size_t idx = (size_t)row * ldc + col;
                    float v = a[j];
                    if (residual) v += residual[idx];
                    C[idx] = v;
                }
            }
        }
    }
}

// ----------------------- depthwise conv (K=4) + SiLU -----------------------
__global__ void conv_silu_kernel(const float* __restrict__ proj,
                                 const float* __restrict__ cw,
                                 const float* __restrict__ cb,
                                 float* __restrict__ xconv) {
    int idx = blockIdx.x * 256 + threadIdx.x;   // over M_TOK * D_INNER
    int c = idx & (D_INNER - 1);
    int m = idx >> 11;
    int t = m & (SEQ_L - 1);
    int b = m >> 11;
    float acc = cb[c];
#pragma unroll
    for (int k = 0; k < 4; ++k) {
        int tk = t - 3 + k;
        if (tk >= 0)
            acc += cw[c * 4 + k] *
                   proj[(size_t)(b * SEQ_L + tk) * D_PROJ + D_INNER + c];
    }
    xconv[idx] = siluf(acc);
}

// --------------------------- dt / dA precompute ----------------------------
__global__ void dt_kernel(const float* __restrict__ proj,
                          const float* __restrict__ dt_bias,
                          const float* __restrict__ A_log,
                          float* __restrict__ dt_buf,
                          float* __restrict__ dA_buf) {
    int idx = blockIdx.x * 256 + threadIdx.x;   // over M_TOK * N_HEADS
    int h = idx & (N_HEADS - 1);
    int m = idx >> 5;
    float raw = proj[(size_t)m * D_PROJ + (D_PROJ - N_HEADS) + h] + dt_bias[h];
    float dt = (raw > 20.f) ? raw : log1pf(expf(raw));
    dt_buf[idx] = dt;
    dA_buf[idx] = expf(-expf(A_log[h]) * dt);
}

// ----------------------------- selective scan ------------------------------
// one block per (batch, head): state h[64][128] in registers (32 f32/thread)
__global__ __launch_bounds__(256) void scan_kernel(
    const float* __restrict__ proj, const float* __restrict__ xconv,
    const float* __restrict__ dt_buf, const float* __restrict__ dA_buf,
    const float* __restrict__ Dvec, float* __restrict__ y) {
    int bh = blockIdx.x;
    int b = bh >> 5;
    int hidx = bh & 31;
    int tid = threadIdx.x;
    int p = tid >> 2;            // 0..63 head-dim element
    int n0 = (tid & 3) * 32;     // state segment

    __shared__ float Bs[D_STATE];
    __shared__ float Cs[D_STATE];
    __shared__ float dtxs[HEAD_DIM];
    __shared__ float part[256];

    float h[32];
#pragma unroll
    for (int j = 0; j < 32; ++j) h[j] = 0.f;
    float Dh = Dvec[hidx];

    for (int t = 0; t < SEQ_L; ++t) {
        int row = b * SEQ_L + t;
        const float* prow = proj + (size_t)row * D_PROJ;
        if (tid < 128)      Bs[tid]       = prow[2 * D_INNER + tid];
        else                Cs[tid - 128] = prow[2 * D_INNER + D_STATE + (tid - 128)];
        if (tid < HEAD_DIM)
            dtxs[tid] = dt_buf[row * N_HEADS + hidx] *
                        xconv[(size_t)row * D_INNER + hidx * HEAD_DIM + tid];
        __syncthreads();

        float dAv = dA_buf[row * N_HEADS + hidx];
        float dtx = dtxs[p];
        float yp = 0.f;
#pragma unroll
        for (int j = 0; j < 32; ++j) {
            h[j] = h[j] * dAv + dtx * Bs[n0 + j];
            yp += h[j] * Cs[n0 + j];
        }
        part[tid] = yp;
        __syncthreads();

        if (tid < HEAD_DIM) {
            float s = part[tid * 4] + part[tid * 4 + 1] +
                      part[tid * 4 + 2] + part[tid * 4 + 3];
            size_t oi = (size_t)row * D_INNER + hidx * HEAD_DIM + tid;
            y[oi] = s + Dh * xconv[oi];
        }
    }
}

// ----------------------------- SiLU(z) gating ------------------------------
__global__ void gate_kernel(const float* __restrict__ proj,
                            const float* __restrict__ y,
                            unsigned short* __restrict__ ybf) {
    int idx = blockIdx.x * 256 + threadIdx.x;   // over M_TOK * D_INNER
    int c = idx & (D_INNER - 1);
    int m = idx >> 11;
    float z = proj[(size_t)m * D_PROJ + c];
    ybf[idx] = f2bfu(y[idx] * siluf(z));
}

// ------------------------------- launcher ----------------------------------
extern "C" void kernel_launch(void* const* d_in, const int* in_sizes, int n_in,
                              void* d_out, int out_size, void* d_ws, size_t ws_size,
                              hipStream_t stream) {
    const float* x        = (const float*)d_in[0];
    const float* norm_w   = (const float*)d_in[1];
    const float* in_proj  = (const float*)d_in[2];
    const float* conv_w   = (const float*)d_in[3];
    const float* conv_b   = (const float*)d_in[4];
    const float* A_log    = (const float*)d_in[5];
    const float* Dvec     = (const float*)d_in[6];
    const float* dt_bias  = (const float*)d_in[7];
    const float* out_proj = (const float*)d_in[8];
    float* out = (float*)d_out;

    char* ws = (char*)d_ws;
    unsigned short* xn_bf = (unsigned short*)ws; ws += (size_t)M_TOK * D_MODEL * 2;
    unsigned short* w1_bf = (unsigned short*)ws; ws += (size_t)D_PROJ * D_MODEL * 2;
    unsigned short* w2_bf = (unsigned short*)ws; ws += (size_t)D_MODEL * D_INNER * 2;
    float* proj  = (float*)ws;                   ws += (size_t)M_TOK * D_PROJ * 4;
    float* xconv = (float*)ws;                   ws += (size_t)M_TOK * D_INNER * 4;
    float* dtb   = (float*)ws;                   ws += (size_t)M_TOK * N_HEADS * 4;
    float* dab   = (float*)ws;                   ws += (size_t)M_TOK * N_HEADS * 4;
    float* ybuf  = (float*)ws;                   ws += (size_t)M_TOK * D_INNER * 4;
    unsigned short* ygate = (unsigned short*)ws; ws += (size_t)M_TOK * D_INNER * 2;
    (void)ws_size; (void)in_sizes; (void)n_in; (void)out_size;

    // weight conversion (re-done each launch: deterministic, capture-safe)
    f32_to_bf16_kernel<<<(D_PROJ * D_MODEL + 255) / 256, 256, 0, stream>>>(
        in_proj, w1_bf, D_PROJ * D_MODEL);
    f32_to_bf16_kernel<<<(D_MODEL * D_INNER + 255) / 256, 256, 0, stream>>>(
        out_proj, w2_bf, D_MODEL * D_INNER);

    rmsnorm_kernel<<<M_TOK, 256, 0, stream>>>(x, norm_w, xn_bf);

    // in_proj: (4096 x 1024) x (1024 x 4384)
    wmma_gemm_kernel<<<dim3((D_PROJ + 127) / 128, M_TOK / 128), 256, 0, stream>>>(
        xn_bf, w1_bf, proj, nullptr, M_TOK, D_PROJ, D_MODEL, D_PROJ);

    conv_silu_kernel<<<(M_TOK * D_INNER) / 256, 256, 0, stream>>>(
        proj, conv_w, conv_b, xconv);

    dt_kernel<<<(M_TOK * N_HEADS) / 256, 256, 0, stream>>>(
        proj, dt_bias, A_log, dtb, dab);

    scan_kernel<<<BATCH * N_HEADS, 256, 0, stream>>>(
        proj, xconv, dtb, dab, Dvec, ybuf);

    gate_kernel<<<(M_TOK * D_INNER) / 256, 256, 0, stream>>>(proj, ybuf, ygate);

    // out_proj: (4096 x 2048) x (2048 x 1024) + residual
    wmma_gemm_kernel<<<dim3(D_MODEL / 128, M_TOK / 128), 256, 0, stream>>>(
        ygate, w2_bf, out, x, M_TOK, D_MODEL, D_INNER, D_MODEL);
}
